// QuantLinear_24721831756587
// MI455X (gfx1250) — matrix operand order repaired and checked by reading, not compiled, not run on hardware
//
#include <hip/hip_runtime.h>

typedef __attribute__((ext_vector_type(16))) _Float16 v16h;
typedef __attribute__((ext_vector_type(8)))  _Float16 v8h;
typedef __attribute__((ext_vector_type(2)))  _Float16 h2;
typedef __attribute__((ext_vector_type(8)))  float    v8f;
typedef __attribute__((ext_vector_type(4)))  int      v4i;

#define OUT_F   11008
#define IN_F    4096
#define MROWS   8
#define MPAD    16                   // pad M to WMMA tile height; rows 8..15 zero
#define KSPLIT  4
#define KCHUNK  (IN_F / KSPLIT)      // 1024
#define NSTEP   (KCHUNK / 32)        // 32 WMMA steps per wave
#define WAVES   8
#define TILE_N  16
#define BLOCK_N (WAVES * TILE_N)     // 128 output features per block

__global__ void zero_f32_kernel(float* __restrict__ p, int n) {
  int i = blockIdx.x * blockDim.x + threadIdx.x;
  if (i < n) p[i] = 0.0f;
}

__global__ __launch_bounds__(256) void q4_wmma_gemm_kernel(
    const float* __restrict__ x,        // [8, 4096] fp32
    const int*   __restrict__ packed,   // [NUMEL/2] int32, one q4_0 byte each
    const float* __restrict__ scales,   // [NUMEL/32] fp32
    float*       __restrict__ out)      // [8, 11008] fp32 (pre-zeroed)
{
  // x K-slice staged as f16, zero-padded to 16 rows: 16 x 1024 x 2B = 32 KB.
  // Padding makes A-fragment loads branchless (no per-step EXEC masking).
  __shared__ alignas(16) _Float16 xs[MPAD * KCHUNK];

  const int tid  = threadIdx.x;
  const int kblk = blockIdx.y * KCHUNK;

  // ---- stage x slice (fp32 -> fp16) into LDS; rows 8..15 zero-filled ----
  for (int i = tid * 4; i < MPAD * KCHUNK; i += 256 * 4) {
    const int r = i / KCHUNK;
    const int c = i - r * KCHUNK;
    if (r < MROWS) {
      const float4 v = *(const float4*)(x + r * IN_F + kblk + c);
      xs[i + 0] = (_Float16)v.x;
      xs[i + 1] = (_Float16)v.y;
      xs[i + 2] = (_Float16)v.z;
      xs[i + 3] = (_Float16)v.w;
    } else {
      xs[i + 0] = (_Float16)0.0f;
      xs[i + 1] = (_Float16)0.0f;
      xs[i + 2] = (_Float16)0.0f;
      xs[i + 3] = (_Float16)0.0f;
    }
  }
  __syncthreads();

  const int wave = tid >> 5;
  const int lane = tid & 31;
  const int row  = lane & 15;   // A-matrix row for this lane (M = lane % 16)
  const int hi   = lane >> 4;   // which K-half of the fragment this lane holds
  const int o    = blockIdx.x * BLOCK_N + wave * TILE_N + row; // output feature

  // Per-lane base pointers (step s adds s*32 halfs / s*16 ints / s scales)
  const _Float16* ap    = xs + row * KCHUNK + hi * 16;
  const int*      wbase = packed + o * (IN_F / 2) + (kblk >> 1) + hi * 8;
  const float*    srow  = scales + o * (IN_F / 32) + (kblk >> 5);

  // ---- software pipeline: prologue loads for step 0 ----
  v8h  alo = *(const v8h*)(ap);
  v8h  ahh = *(const v8h*)(ap + 8);
  v4i  w0  = *(const v4i*)(wbase);
  v4i  w1  = *(const v4i*)(wbase + 4);
  float s32 = srow[0];

  v8f acc = {};

#pragma unroll 2
  for (int s = 0; s < NSTEP; ++s) {
    // capture current step's operands
    const v8h  c_lo = alo;
    const v8h  c_hh = ahh;
    const v4i  cw0  = w0;
    const v4i  cw1  = w1;
    const float cs  = s32;

    // issue next step's loads early (clamped on last step; harmless reload)
    const int sn = (s + 1 < NSTEP) ? (s + 1) : s;
    const _Float16* pn  = ap + sn * 32;
    const int*      wpn = wbase + sn * 16;
    alo = *(const v8h*)(pn);
    ahh = *(const v8h*)(pn + 8);
    w0  = *(const v4i*)(wpn);
    w1  = *(const v4i*)(wpn + 4);
    s32 = srow[sn];
    __builtin_prefetch(wpn + 256, 0, 1);   // stream ahead on the weight row

    // ---- assemble A fragment (contiguous k = 16*hi + i, same order as B) ----
    v16h a;
#pragma unroll
    for (int i = 0; i < 8; ++i) { a[i] = c_lo[i]; a[8 + i] = c_hh[i]; }

    // ---- dequantize 16 q4_0 weights -> B fragment ----
    const _Float16 sh = (_Float16)cs;
    const _Float16 bh = (_Float16)(-1032.0f * cs);
    const h2 s2 = {sh, sh};
    const h2 b2 = {bh, bh};

    v16h b;
#pragma unroll
    for (int i = 0; i < 8; ++i) {
      const unsigned q = (unsigned)((i < 4) ? cw0[i] : cw1[i - 4]);
      // pack (low,high) nibbles as f16 (1024+n) pair; (n-8)*s via one pk_fma
      const unsigned u = ((q | (q << 12)) & 0x000F000Fu) | 0x64006400u;
      union { unsigned u32; h2 h; } cv;
      cv.u32 = u;
      const h2 wv = cv.h * s2 + b2;        // s*(1024+n) - 1032*s = (n-8)*s
      b[2 * i]     = wv[0];                // element 2i   = low nibble
      b[2 * i + 1] = wv[1];                // element 2i+1 = high nibble
    }

    // 8 args: (neg_a, A, neg_b, B, c_mod, C, reuse_a, reuse_b)
    acc = __builtin_amdgcn_wmma_f32_16x16x32_f16(
        false, a, false, b, (short)0, acc, false, false);
  }

  // ---- C/D layout: lanes 0-15 hold M=0..7 in VGPR 0..7 (valid rows) ----
  if (hi == 0) {
#pragma unroll
    for (int m = 0; m < MROWS; ++m)
      atomicAdd(out + m * OUT_F + o, acc[m]);
  }
}

extern "C" void kernel_launch(void* const* d_in, const int* in_sizes, int n_in,
                              void* d_out, int out_size, void* d_ws, size_t ws_size,
                              hipStream_t stream) {
  const float* x      = (const float*)d_in[0];
  const int*   packed = (const int*)d_in[1];
  const float* scales = (const float*)d_in[2];
  float*       out    = (float*)d_out;

  // out is poisoned before timing; K-split partials are atomically accumulated,
  // so zero it first (graph-capture safe: plain kernel on `stream`).
  zero_f32_kernel<<<(out_size + 255) / 256, 256, 0, stream>>>(out, out_size);

  dim3 grid(OUT_F / BLOCK_N, KSPLIT);
  q4_wmma_gemm_kernel<<<grid, 256, 0, stream>>>(x, packed, scales, out);
}
